// Linear_MoE_8091718385700
// MI455X (gfx1250) — matrix operand order repaired and checked by reading, not compiled, not run on hardware
//
#include <hip/hip_runtime.h>
#include <hip/hip_bf16.h>
#include <math.h>

// ---------------------------------------------------------------- constants
#define N_TOK  8192
#define D_IN   2048
#define D_OUT  2048
#define N_EXP  8
#define TOP_K  2

#define TM 128          // M tile (tokens)
#define TN 128          // N tile (out features)
#define TK 32           // K step (bf16 wmma K)
#define LDT 40          // LDS row stride in elements (padded, 80B -> 16B aligned)
#define MAX_MT (N_TOK / TM)   // 64 max M-tiles per expert

typedef __attribute__((ext_vector_type(16))) __bf16 v16bf;
typedef __attribute__((ext_vector_type(8)))  float  v8f;
typedef __attribute__((ext_vector_type(4)))  float  v4f;
typedef __attribute__((ext_vector_type(4)))  __bf16 v4bf;
typedef unsigned int v4u __attribute__((ext_vector_type(4)));
typedef unsigned int v2u __attribute__((ext_vector_type(2)));

union Frag { v16bf v; v4u q[2]; };

// split a 4-vector into bf16 hi + bf16 lo (packed as 2x u32 for ds_store_b64)
__device__ __forceinline__ void split4(v4f a, v2u& h, v2u& l) {
    v4bf hb = __builtin_convertvector(a, v4bf);
    v4f  hf = __builtin_convertvector(hb, v4f);
    v4bf lb = __builtin_convertvector(a - hf, v4bf);
    union { v4bf b; v2u u; } ph, pl;
    ph.b = hb; pl.b = lb;
    h = ph.u; l = pl.u;
}

// ---------------------------------------------------------------- kernel 0
__global__ void zero_counts_kernel(int* counts) {
    if (threadIdx.x < N_EXP) counts[threadIdx.x] = 0;
}

// ---------------------------------------------------------------- kernel 1
// One wave (32 lanes) per token: fp32 logits, top-2, softmax, routing.
__global__ __launch_bounds__(256)
void gate_kernel(const float* __restrict__ x, const float* __restrict__ gw,
                 const float* __restrict__ gb, float* __restrict__ gates,
                 int* __restrict__ counts, int* __restrict__ perm_tok,
                 float* __restrict__ perm_w, int* __restrict__ topk_idx,
                 float* __restrict__ topk_w) {
    const int wave = threadIdx.x >> 5;
    const int lane = threadIdx.x & 31;
    const int n = blockIdx.x * 8 + wave;

    float acc[N_EXP];
#pragma unroll
    for (int e = 0; e < N_EXP; ++e) acc[e] = 0.f;

    const float* xr = x + (size_t)n * D_IN;
    for (int d = lane; d < D_IN; d += 32) {
        const float xv = xr[d];
        const float* g = gw + (size_t)d * N_EXP;
#pragma unroll
        for (int e = 0; e < N_EXP; ++e) acc[e] += xv * g[e];
    }
    // wave32 butterfly reduction
#pragma unroll
    for (int off = 16; off > 0; off >>= 1) {
#pragma unroll
        for (int e = 0; e < N_EXP; ++e) acc[e] += __shfl_xor(acc[e], off, 32);
    }

    if (lane == 0) {
        float lg[N_EXP];
#pragma unroll
        for (int e = 0; e < N_EXP; ++e) lg[e] = acc[e] + gb[e];
        // top-2, lowest index wins ties (matches jax.lax.top_k)
        int i0 = 0;
#pragma unroll
        for (int e = 1; e < N_EXP; ++e) if (lg[e] > lg[i0]) i0 = e;
        int i1 = (i0 == 0) ? 1 : 0;
#pragma unroll
        for (int e = 0; e < N_EXP; ++e) if (e != i0 && lg[e] > lg[i1]) i1 = e;

        const float v0 = lg[i0], v1 = lg[i1];
        const float e1 = expf(v1 - v0);
        const float inv = 1.f / (1.f + e1);
        const float w0 = inv, w1 = e1 * inv;

        float* gr = gates + (size_t)n * N_EXP;
#pragma unroll
        for (int e = 0; e < N_EXP; ++e) gr[e] = 0.f;
        gr[i0] = w0; gr[i1] = w1;

        topk_idx[n * 2 + 0] = i0;  topk_idx[n * 2 + 1] = i1;
        topk_w[n * 2 + 0]  = w0;  topk_w[n * 2 + 1]  = w1;

        int s0 = atomicAdd(&counts[i0], 1);
        perm_tok[i0 * N_TOK + s0] = n;  perm_w[i0 * N_TOK + s0] = w0;
        int s1 = atomicAdd(&counts[i1], 1);
        perm_tok[i1 * N_TOK + s1] = n;  perm_w[i1 * N_TOK + s1] = w1;
    }
}

// ---------------------------------------------------------------- kernel 2
// out[n,o] = w0*b[e0,o] + w1*b[e1,o]  (fully initializes out before GEMM adds)
__global__ __launch_bounds__(256)
void bias_init_kernel(const float* __restrict__ eb, const int* __restrict__ topk_idx,
                      const float* __restrict__ topk_w, float* __restrict__ out) {
    const size_t gid = (size_t)blockIdx.x * 256 + threadIdx.x;
    const int n = (int)(gid >> 11);          // / D_OUT
    const int o = (int)(gid & (D_OUT - 1));
    const int i0 = topk_idx[n * 2], i1 = topk_idx[n * 2 + 1];
    out[gid] = topk_w[n * 2] * eb[(size_t)i0 * D_OUT + o]
             + topk_w[n * 2 + 1] * eb[(size_t)i1 * D_OUT + o];
}

// ---------------------------------------------------------------- kernel 3
// Grouped expert GEMM. grid.x = expert*64 + mtile, grid.y = ntile.
// A rows = gathered x rows scaled by gate weight, bf16 hi/lo split.
// acc += Ahi*Bhi + Ahi*Blo + Alo*Bhi  (bf16x2 ~ fp32 accuracy), fp32 accum.
// Register double-buffering: next K-tile global loads issue before the WMMA
// block so HBM/L2 latency hides behind matrix math.
__global__ __launch_bounds__(256)
void moe_gemm_kernel(const float* __restrict__ x, const float* __restrict__ ew,
                     float* __restrict__ out, const int* __restrict__ counts,
                     const int* __restrict__ perm_tok, const float* __restrict__ perm_w) {
    const int e  = blockIdx.x >> 6;
    const int mt = blockIdx.x & (MAX_MT - 1);
    const int cnt = counts[e];
    const int m0 = mt * TM;
    if (m0 >= cnt) return;                       // empty tile: early exit
    const int rows = (cnt - m0 < TM) ? (cnt - m0) : TM;
    const int nb = blockIdx.y * TN;

    const float* W = ew + (size_t)e * D_IN * D_OUT;
    const int*   ptok = perm_tok + e * N_TOK + m0;
    const float* pw   = perm_w   + e * N_TOK + m0;

    __shared__ __align__(16) __bf16 Ah[TM * LDT];
    __shared__ __align__(16) __bf16 Al[TM * LDT];
    __shared__ __align__(16) __bf16 Bh[TN * LDT];   // transposed: [n][k]
    __shared__ __align__(16) __bf16 Bl[TN * LDT];

    const int t    = threadIdx.x;
    const int lane = t & 31;
    const int w    = t >> 5;
    const int wm   = w >> 1;        // 0..3  -> 32-row strip
    const int wn   = w & 1;         // 0..1  -> 64-col strip
    const int half = lane >> 4;     // 0/1
    const int mr   = lane & 15;

    const v8f vzero = {0.f, 0.f, 0.f, 0.f, 0.f, 0.f, 0.f, 0.f};
    v8f acc[2][4];
#pragma unroll
    for (int m = 0; m < 2; ++m)
#pragma unroll
        for (int nf = 0; nf < 4; ++nf) acc[m][nf] = vzero;

    // A-stage mapping: thread -> (row, 16-wide k chunk)
    const int ar  = t >> 1;
    const int akh = (t & 1) * 16;
    const bool arow_ok = (ar < rows);
    const int   tokr = arow_ok ? ptok[ar] : 0;
    const float wr   = arow_ok ? pw[ar]   : 0.f;
    const float* xr  = x + (size_t)tokr * D_IN;

    // B-stage mapping: thread -> 4k x 4n micro-tile
    const int bkk = (t & 7) * 4;       // 0,4,...,28
    const int bn0 = (t >> 3) * 4;      // 0,4,...,124
    const float* wbase = W + (size_t)bkk * D_OUT + nb + bn0;

    const v4f f4zero = {0.f, 0.f, 0.f, 0.f};
    v4f aR[4], bR[4];

    // ---- prologue loads (k0 = 0)
#pragma unroll
    for (int i = 0; i < 4; ++i) {
        aR[i] = arow_ok ? (*(const v4f*)(xr + akh + i * 4)) * wr : f4zero;
        bR[i] = *(const v4f*)(wbase + (size_t)i * D_OUT);
    }

    for (int k0 = 0; k0 < D_IN; k0 += TK) {
        // ---- stage A tile (packed b64 stores: 4 bf16 per store)
#pragma unroll
        for (int i = 0; i < 4; ++i) {
            v2u h, l;
            split4(aR[i], h, l);
            *(v2u*)&Ah[ar * LDT + akh + i * 4] = h;
            *(v2u*)&Al[ar * LDT + akh + i * 4] = l;
        }
        // ---- stage B tile transposed (per column: 4 consecutive k packed)
#pragma unroll
        for (int j = 0; j < 4; ++j) {
            v4f col = {bR[0][j], bR[1][j], bR[2][j], bR[3][j]};
            v2u h, l;
            split4(col, h, l);
            *(v2u*)&Bh[(bn0 + j) * LDT + bkk] = h;
            *(v2u*)&Bl[(bn0 + j) * LDT + bkk] = l;
        }
        __syncthreads();

        // ---- issue next K-tile global loads (latency hidden by WMMAs below)
        v4f aN[4], bN[4];
        if (k0 + TK < D_IN) {
            const float* xn = xr + k0 + TK + akh;
            const float* wn_ = wbase + (size_t)(k0 + TK) * D_OUT;
#pragma unroll
            for (int i = 0; i < 4; ++i) {
                aN[i] = arow_ok ? (*(const v4f*)(xn + i * 4)) * wr : f4zero;
                bN[i] = *(const v4f*)(wn_ + (size_t)i * D_OUT);
            }
        } else {
#pragma unroll
            for (int i = 0; i < 4; ++i) { aN[i] = f4zero; bN[i] = f4zero; }
        }

        // ---- fragment loads per ISA 16-bit A/B layouts (two b128 each)
        Frag ah[2], al[2];
#pragma unroll
        for (int m = 0; m < 2; ++m) {
            const __bf16* pA = Ah + (wm * 32 + m * 16 + mr) * LDT + half * 8;
            ah[m].q[0] = *(const v4u*)(pA);
            ah[m].q[1] = *(const v4u*)(pA + 16);
            const __bf16* pL = Al + (wm * 32 + m * 16 + mr) * LDT + half * 8;
            al[m].q[0] = *(const v4u*)(pL);
            al[m].q[1] = *(const v4u*)(pL + 16);
        }
#pragma unroll
        for (int nf = 0; nf < 4; ++nf) {
            const int col = wn * 64 + nf * 16 + mr;
            const __bf16* pB = Bh + col * LDT + half * 8;
            Frag bh; bh.q[0] = *(const v4u*)(pB); bh.q[1] = *(const v4u*)(pB + 16);
            const __bf16* pBl = Bl + col * LDT + half * 8;
            Frag bl; bl.q[0] = *(const v4u*)(pBl); bl.q[1] = *(const v4u*)(pBl + 16);
#pragma unroll
            for (int m = 0; m < 2; ++m) {
                acc[m][nf] = __builtin_amdgcn_wmma_f32_16x16x32_bf16(
                    false, ah[m].v, false, bh.v, (short)0, acc[m][nf], false, false);
                acc[m][nf] = __builtin_amdgcn_wmma_f32_16x16x32_bf16(
                    false, ah[m].v, false, bl.v, (short)0, acc[m][nf], false, false);
                acc[m][nf] = __builtin_amdgcn_wmma_f32_16x16x32_bf16(
                    false, al[m].v, false, bh.v, (short)0, acc[m][nf], false, false);
            }
        }
        __syncthreads();

#pragma unroll
        for (int i = 0; i < 4; ++i) { aR[i] = aN[i]; bR[i] = bN[i]; }
    }

    // ---- epilogue: scatter-add (token rows may also appear under other expert)
#pragma unroll
    for (int m = 0; m < 2; ++m) {
        const int rbase = wm * 32 + m * 16 + half * 8;
#pragma unroll
        for (int j = 0; j < 8; ++j) {
            const int r = rbase + j;
            if (r < rows) {
                const int tok = ptok[r];
                float* orow = out + (size_t)tok * D_OUT + nb + wn * 64 + mr;
#pragma unroll
                for (int nf = 0; nf < 4; ++nf)
                    atomicAdd(orow + nf * 16, acc[m][nf][j]);
            }
        }
    }
}

// ---------------------------------------------------------------- launcher
extern "C" void kernel_launch(void* const* d_in, const int* in_sizes, int n_in,
                              void* d_out, int out_size, void* d_ws, size_t ws_size,
                              hipStream_t stream) {
    const float* x  = (const float*)d_in[0];
    const float* gw = (const float*)d_in[1];
    const float* gb = (const float*)d_in[2];
    const float* ew = (const float*)d_in[3];
    const float* eb = (const float*)d_in[4];

    float* out   = (float*)d_out;
    float* gates = out + (size_t)N_TOK * D_OUT;

    // workspace layout
    int*   counts   = (int*)d_ws;                       // 16 ints (8 used)
    int*   perm_tok = counts + 16;                      // 8*8192
    float* perm_w   = (float*)(perm_tok + N_EXP * N_TOK);
    int*   topk_idx = (int*)(perm_w + N_EXP * N_TOK);   // 8192*2
    float* topk_w   = (float*)(topk_idx + N_TOK * 2);   // 8192*2

    zero_counts_kernel<<<1, 32, 0, stream>>>(counts);
    gate_kernel<<<N_TOK / 8, 256, 0, stream>>>(x, gw, gb, gates, counts,
                                               perm_tok, perm_w, topk_idx, topk_w);
    bias_init_kernel<<<(N_TOK * (size_t)D_OUT) / 256, 256, 0, stream>>>(
        eb, topk_idx, topk_w, out);
    moe_gemm_kernel<<<dim3(N_EXP * MAX_MT, D_OUT / TN), 256, 0, stream>>>(
        x, ew, out, counts, perm_tok, perm_w);
}
